// MoE_66039417143340
// MI455X (gfx1250) — matrix operand order repaired and checked by reading, not compiled, not run on hardware
//
#include <hip/hip_runtime.h>
#include <hip/hip_bf16.h>

// ---------------- problem constants (match reference) ----------------
constexpr int B_TOK   = 8192;
constexpr int IN_DIM  = 512;
constexpr int HIDDEN  = 1024;
constexpr int OUT_DIM = 512;
constexpr int NEXP    = 8;

// ---------------- workspace layout (bytes) ----------------
constexpr size_t WS_COUNTS  = 0;                                   // 8 ints
constexpr size_t WS_BUCKETS = 1024;                                // 8*8192 ints
constexpr size_t WS_XB      = WS_BUCKETS + (size_t)NEXP * B_TOK * 4;          // bf16 [B][512]
constexpr size_t WS_W1T     = WS_XB  + (size_t)B_TOK * IN_DIM * 2;            // bf16 [E][HIDDEN][IN_DIM]
constexpr size_t WS_W2T     = WS_W1T + (size_t)NEXP * HIDDEN * IN_DIM * 2;    // bf16 [E][OUT_DIM][HIDDEN]

// ---------------- CDNA5 WMMA vector types ----------------
typedef __attribute__((ext_vector_type(16))) __bf16 v16bf;
typedef __attribute__((ext_vector_type(8)))  __bf16 v8bf;
typedef __attribute__((ext_vector_type(8)))  float  v8f;

union V16 { v16bf v; v8bf h[2]; };

__device__ __forceinline__ unsigned short f32_to_bf16(float f) {
    unsigned int u = __float_as_uint(f);
    u += 0x7FFFu + ((u >> 16) & 1u);     // round-to-nearest-even
    return (unsigned short)(u >> 16);
}

// ---------------- kernel 0: zero the per-expert counters ----------------
__global__ void moe_zero_counts(int* counts) {
    if (threadIdx.x < NEXP) counts[threadIdx.x] = 0;
}

// ---------------- kernel 1: x fp32 -> bf16 (coalesced) ----------------
__global__ void moe_convert_x(const float* __restrict__ x,
                              unsigned short* __restrict__ xb) {
    const int total = B_TOK * IN_DIM;
    for (int i = blockIdx.x * blockDim.x + threadIdx.x; i < total;
         i += gridDim.x * blockDim.x)
        xb[i] = f32_to_bf16(x[i]);
}

// --------- kernel 2: W [E][K][N] fp32 -> Wt [E][N][K] bf16 (transpose) ---------
template <int K, int N>
__global__ void moe_convert_wT(const float* __restrict__ W,
                               unsigned short* __restrict__ Wt) {
    constexpr int total = NEXP * K * N;
    for (int i = blockIdx.x * blockDim.x + threadIdx.x; i < total;
         i += gridDim.x * blockDim.x) {
        int e = i / (K * N);
        int r = i - e * (K * N);
        int n = r / K;          // output index i = ((e*N)+n)*K + k  (k contiguous)
        int k = r - n * K;
        Wt[i] = f32_to_bf16(W[((size_t)e * K + k) * N + n]);
    }
}

// ---------------- kernel 3: gating + top-1 routing ----------------
// one wave (32 lanes) per token
__global__ void moe_gating(const float* __restrict__ x,
                           const float* __restrict__ Wg,
                           const float* __restrict__ bg,
                           int* __restrict__ counts,
                           int* __restrict__ buckets) {
    int wave = threadIdx.x >> 5;
    int lane = threadIdx.x & 31;
    int token = blockIdx.x * (blockDim.x >> 5) + wave;
    if (token >= B_TOK) return;

    const float* xr = x + (size_t)token * IN_DIM;
    float acc[NEXP];
#pragma unroll
    for (int e = 0; e < NEXP; ++e) acc[e] = 0.f;

    for (int k = lane; k < IN_DIM; k += 32) {
        float xv = xr[k];
        const float* wr = Wg + k * NEXP;
#pragma unroll
        for (int e = 0; e < NEXP; ++e) acc[e] += xv * wr[e];
    }
#pragma unroll
    for (int e = 0; e < NEXP; ++e) {
#pragma unroll
        for (int off = 16; off > 0; off >>= 1)
            acc[e] += __shfl_xor(acc[e], off, 32);
    }
    if (lane == 0) {
        int best = 0;
        float bv = acc[0] + bg[0];
#pragma unroll
        for (int e = 1; e < NEXP; ++e) {
            float v = acc[e] + bg[e];
            if (v > bv) { bv = v; best = e; }   // first-max tie-break like argmax
        }
        int pos = atomicAdd(&counts[best], 1);
        buckets[best * B_TOK + pos] = token;
    }
}

// ---------------- kernel 4: fused routed 2-layer expert MLP ----------------
// 256 threads = 8 waves; one block = one expert's 32-token tile.
// Waves split 2 (M groups of 16 rows) x 4 (N groups).
// Stage A: h[32 x 1024] = relu(Xe @ W1 + b1) -> bf16 in LDS (64 KB)
// Stage B: y[32 x 512]  = h @ W2 + b2 -> scatter f32 rows to out[token]
__global__ __launch_bounds__(256)
void moe_mlp(const unsigned short* __restrict__ xb,
             const unsigned short* __restrict__ W1t,   // [E][HIDDEN][IN_DIM] bf16
             const unsigned short* __restrict__ W2t,   // [E][OUT_DIM][HIDDEN] bf16
             const float* __restrict__ b1,
             const float* __restrict__ b2,
             const int* __restrict__ counts,
             const int* __restrict__ buckets,
             float* __restrict__ out) {
    constexpr int MT = 32;
    constexpr int TILES_MAX = B_TOK / MT;   // 256

    int e    = blockIdx.x / TILES_MAX;
    int tile = blockIdx.x - e * TILES_MAX;
    int cnt  = counts[e];
    int ntiles = (cnt + MT - 1) / MT;
    if (tile >= ntiles) return;             // uniform per block

    __shared__ __align__(16) unsigned short h_lds[MT * HIDDEN];   // 64 KB
    __shared__ int tok_lds[MT];

    const int* bucket = buckets + e * B_TOK;
    if (threadIdx.x < MT) {
        int m = tile * MT + (int)threadIdx.x;
        tok_lds[threadIdx.x] = bucket[m < cnt ? m : cnt - 1];     // clamp pad rows
    }
    __syncthreads();

    const int lane = threadIdx.x & 31;
    const int wave = threadIdx.x >> 5;
    const int mg   = wave >> 2;        // 0..1 : which 16-row group
    const int ng   = wave & 3;         // 0..3 : which N group
    const int hv   = lane >> 4;        // half-select (K sub-block)
    const int lcol = lane & 15;        // column within 16x16 tile / A row

    // ---------------- Stage A : 16 rows x 256 cols per wave ----------------
    const unsigned short* arow = xb + (size_t)tok_lds[mg * 16 + lcol] * IN_DIM;
    const unsigned short* w1e  = W1t + (size_t)e * HIDDEN * IN_DIM;

    v8f acc[16];
#pragma unroll
    for (int j = 0; j < 16; ++j) acc[j] = (v8f){0.f,0.f,0.f,0.f,0.f,0.f,0.f,0.f};

    for (int k0 = 0; k0 < IN_DIM; k0 += 32) {
        V16 a;                               // A frag: K = k0 + 16*hv + [0..15]
        a.h[0] = *(const v8bf*)(arow + k0 + 16 * hv);
        a.h[1] = *(const v8bf*)(arow + k0 + 16 * hv + 8);
#pragma unroll
        for (int j = 0; j < 16; ++j) {
            int col = ng * 256 + j * 16 + lcol;
            const unsigned short* brow = w1e + (size_t)col * IN_DIM + k0 + 16 * hv;
            V16 b;                           // B frag: K = k0 + 16*hv + [0..15]
            b.h[0] = *(const v8bf*)(brow);
            b.h[1] = *(const v8bf*)(brow + 8);
            acc[j] = __builtin_amdgcn_wmma_f32_16x16x32_bf16(
                false, a.v, false, b.v, (short)0, acc[j], false, false);
        }
    }
    // bias + relu -> bf16 LDS (C layout: vgpr i -> row i + 8*hv, col = lcol)
#pragma unroll
    for (int j = 0; j < 16; ++j) {
        int col = ng * 256 + j * 16 + lcol;
        float bias = b1[e * HIDDEN + col];
#pragma unroll
        for (int i = 0; i < 8; ++i) {
            int m = mg * 16 + i + 8 * hv;
            float v = acc[j][i] + bias;
            v = v > 0.f ? v : 0.f;
            h_lds[m * HIDDEN + col] = f32_to_bf16(v);
        }
    }
    __syncthreads();

    // ---------------- Stage B : 16 rows x 128 cols per wave ----------------
    const unsigned short* hrow = h_lds + (size_t)(mg * 16 + lcol) * HIDDEN;
    const unsigned short* w2e  = W2t + (size_t)e * OUT_DIM * HIDDEN;

    v8f acc2[8];
#pragma unroll
    for (int j = 0; j < 8; ++j) acc2[j] = (v8f){0.f,0.f,0.f,0.f,0.f,0.f,0.f,0.f};

    for (int k0 = 0; k0 < HIDDEN; k0 += 32) {
        V16 a;
        a.h[0] = *(const v8bf*)(hrow + k0 + 16 * hv);
        a.h[1] = *(const v8bf*)(hrow + k0 + 16 * hv + 8);
#pragma unroll
        for (int j = 0; j < 8; ++j) {
            int col = ng * 128 + j * 16 + lcol;
            const unsigned short* brow = w2e + (size_t)col * HIDDEN + k0 + 16 * hv;
            V16 b;
            b.h[0] = *(const v8bf*)(brow);
            b.h[1] = *(const v8bf*)(brow + 8);
            acc2[j] = __builtin_amdgcn_wmma_f32_16x16x32_bf16(
                false, a.v, false, b.v, (short)0, acc2[j], false, false);
        }
    }
    // bias + scatter valid rows to out[token]
#pragma unroll
    for (int j = 0; j < 8; ++j) {
        int col = ng * 128 + j * 16 + lcol;
        float bias = b2[e * OUT_DIM + col];
#pragma unroll
        for (int i = 0; i < 8; ++i) {
            int m  = mg * 16 + i + 8 * hv;
            int gm = tile * MT + m;
            if (gm < cnt) {
                out[(size_t)tok_lds[m] * OUT_DIM + col] = acc2[j][i] + bias;
            }
        }
    }
}

// ---------------- host-side launcher ----------------
extern "C" void kernel_launch(void* const* d_in, const int* in_sizes, int n_in,
                              void* d_out, int out_size, void* d_ws, size_t ws_size,
                              hipStream_t stream) {
    const float* x  = (const float*)d_in[0];
    const float* Wg = (const float*)d_in[1];
    const float* bg = (const float*)d_in[2];
    const float* W1 = (const float*)d_in[3];
    const float* b1 = (const float*)d_in[4];
    const float* W2 = (const float*)d_in[5];
    const float* b2 = (const float*)d_in[6];
    float* out = (float*)d_out;

    char* ws = (char*)d_ws;
    int* counts  = (int*)(ws + WS_COUNTS);
    int* buckets = (int*)(ws + WS_BUCKETS);
    unsigned short* xb  = (unsigned short*)(ws + WS_XB);
    unsigned short* w1t = (unsigned short*)(ws + WS_W1T);
    unsigned short* w2t = (unsigned short*)(ws + WS_W2T);

    moe_zero_counts<<<1, 32, 0, stream>>>(counts);
    moe_convert_x<<<2048, 256, 0, stream>>>(x, xb);
    moe_convert_wT<IN_DIM, HIDDEN><<<2048, 256, 0, stream>>>(W1, w1t);
    moe_convert_wT<HIDDEN, OUT_DIM><<<2048, 256, 0, stream>>>(W2, w2t);
    moe_gating<<<B_TOK / 8, 256, 0, stream>>>(x, Wg, bg, counts, buckets);
    moe_mlp<<<NEXP * (B_TOK / 32), 256, 0, stream>>>(xb, w1t, w2t, b1, b2,
                                                     counts, buckets, out);
}
